// DecoderStructural_59021440581735
// MI455X (gfx1250) — matrix operand order repaired and checked by reading, not compiled, not run on hardware
//
#include <hip/hip_runtime.h>

typedef __attribute__((ext_vector_type(16))) _Float16 v16h;
typedef __attribute__((ext_vector_type(8)))  _Float16 v8h;
typedef __attribute__((ext_vector_type(8)))  float    v8f;

#define B_   64
#define P_   256
#define ENC_ 512
#define HID_ 512
#define ATT_ 512
#define EMB_ 256
#define VOC_ 32
#define T_   100

// ---------------- conversion helpers (one-time per call) ----------------

__global__ void k_convert_f16(const float* __restrict__ in, _Float16* __restrict__ out, int n) {
    int i = blockIdx.x * 256 + threadIdx.x;
    if (i < n) out[i] = (_Float16)in[i];
}

// in: (K rows x N cols) f32 row-major -> out: (N rows x K cols) f16 row-major
__global__ void k_transpose_f16(const float* __restrict__ in, _Float16* __restrict__ out, int K, int N) {
    int i = blockIdx.x * 256 + threadIdx.x;
    if (i >= K * N) return;
    int k = i / N, n = i % N;
    out[(size_t)n * K + k] = (_Float16)in[i];
}

__global__ void k_init(float* __restrict__ h, _Float16* __restrict__ h16, float* __restrict__ loss) {
    int i = blockIdx.x * 256 + threadIdx.x;
    if (i < B_ * HID_) { h[i] = 0.0f; h16[i] = (_Float16)0.0f; }
    if (i == 0) *loss = 0.0f;
}

// ---------------- WMMA GEMM: C[M,N] = A16[M,K] * Bt16[N,K]^T + bias[N] ----
// A row-major f16 (lda=K), Bt holds B column-major (row n = column n of B).
// One wave computes one 16x16 tile; K-loop in chunks of 32 via
// v_wmma_f32_16x16x32_f16 (fp32 accumulation).

__global__ __launch_bounds__(256) void k_wmma_gemm(
    const _Float16* __restrict__ A, const _Float16* __restrict__ Bt,
    const float* __restrict__ bias, float* __restrict__ C,
    int M, int N, int K)
{
    const int lane = threadIdx.x & 31;
    const int wave = threadIdx.x >> 5;
    const int tilesN = N >> 4;
    const int totalTiles = (M >> 4) * tilesN;
    const int tile = blockIdx.x * 8 + wave;
    if (tile >= totalTiles) return;          // wave-uniform: EXEC all-1 for WMMA

    const int tm  = (tile / tilesN) << 4;
    const int tn  = (tile % tilesN) << 4;
    const int r16 = lane & 15;               // A row / B col / C col index
    const int hi  = lane >> 4;

    const _Float16* aRow = A  + (size_t)(tm + r16) * K;
    const _Float16* bRow = Bt + (size_t)(tn + r16) * K + hi * 16;

    v8f c = {};
    for (int k0 = 0; k0 < K; k0 += 32) {
        // A 16x32 fragment: lane holds row (lane&15); K = k0 + hi*16 + j (contig 16)
        v16h a = *(const v16h*)(aRow + k0 + hi * 16);
        // B 32x16 fragment: lane holds col (lane&15); K = k0 + hi*16 + j (contig 16)
        v16h b = *(const v16h*)(bRow + k0);
        c = __builtin_amdgcn_wmma_f32_16x16x32_f16(false, a, false, b,
                                                   (short)0, c, false, false);
    }
    const float bv = bias[tn + r16];
    float* cPtr = C + (size_t)(tm + hi * 8) * N + tn + r16;
#pragma unroll
    for (int r = 0; r < 8; ++r)
        cPtr[(size_t)r * N] = c[r] + bv;     // C layout: M = r + 8*hi, N = lane&15
}

// ---------------- attention scores: s[b,p] = sum_a v[a]*tanh(encp[b,p,a]+g[b,a])

__global__ __launch_bounds__(256) void k_att_scores(
    const float* __restrict__ encp, const float* __restrict__ g,
    const float* __restrict__ vatt, float* __restrict__ scores)
{
    __shared__ float gs[ATT_];
    __shared__ float vs[ATT_];
    const int b = blockIdx.x, tid = threadIdx.x;
    gs[tid]       = g[b * ATT_ + tid];
    gs[tid + 256] = g[b * ATT_ + tid + 256];
    vs[tid]       = vatt[tid];
    vs[tid + 256] = vatt[tid + 256];
    __syncthreads();
    const int lane = tid & 31, wave = tid >> 5;
    for (int p = wave; p < P_; p += 8) {
        const float* row = encp + ((size_t)b * P_ + p) * ATT_;
        float acc = 0.0f;
        for (int a = lane; a < ATT_; a += 32)
            acc += vs[a] * tanhf(row[a] + gs[a]);
        for (int off = 16; off; off >>= 1) acc += __shfl_xor(acc, off, 32);
        if (lane == 0) scores[b * P_ + p] = acc;
    }
}

// ---------------- softmax over P, ctx = alpha @ Ef, build x16 = [ctx | emb] ---

__global__ __launch_bounds__(256) void k_softmax_ctx(
    const float* __restrict__ scores, const float* __restrict__ Ef,
    const _Float16* __restrict__ emb16, const int* __restrict__ tgt,
    _Float16* __restrict__ x16, int t)
{
    __shared__ float al[P_];
    __shared__ float red[256];
    const int b = blockIdx.x, tid = threadIdx.x;
    float sv = scores[b * P_ + tid];
    red[tid] = sv; __syncthreads();
    for (int s = 128; s; s >>= 1) { if (tid < s) red[tid] = fmaxf(red[tid], red[tid + s]); __syncthreads(); }
    float m = red[0]; __syncthreads();
    float e = __expf(sv - m);
    red[tid] = e; __syncthreads();
    for (int s = 128; s; s >>= 1) { if (tid < s) red[tid] += red[tid + s]; __syncthreads(); }
    float inv = 1.0f / red[0];
    al[tid] = e * inv;
    __syncthreads();

    const float* base = Ef + (size_t)b * P_ * ENC_;
    float acc0 = 0.0f, acc1 = 0.0f;
    for (int p = 0; p < P_; ++p) {
        float a = al[p];
        acc0 += a * base[(size_t)p * ENC_ + tid];
        acc1 += a * base[(size_t)p * ENC_ + tid + 256];
    }
    _Float16* xb = x16 + (size_t)b * (ENC_ + EMB_);
    xb[tid]       = (_Float16)acc0;
    xb[tid + 256] = (_Float16)acc1;
    int tok = (t == 0) ? 0 : tgt[b * T_ + (t - 1)];   // START = 0
    xb[ENC_ + tid] = emb16[tok * EMB_ + tid];
}

// ---------------- GRU elementwise + state/storage writeback -------------------

__global__ __launch_bounds__(256) void k_gru(
    const float* __restrict__ gi, const float* __restrict__ gh,
    float* __restrict__ h, _Float16* __restrict__ h16,
    float* __restrict__ storage)
{
    int i = blockIdx.x * 256 + threadIdx.x;        // < B*HID
    int b = i >> 9, j = i & (HID_ - 1);
    const float* gib = gi + (size_t)b * 3 * HID_;
    const float* ghb = gh + (size_t)b * 3 * HID_;
    float ir = gib[j], iz = gib[HID_ + j], inn = gib[2 * HID_ + j];
    float hr = ghb[j], hz = ghb[HID_ + j], hnn = ghb[2 * HID_ + j];
    float r = 1.0f / (1.0f + __expf(-(ir + hr)));
    float z = 1.0f / (1.0f + __expf(-(iz + hz)));
    float n = tanhf(inn + r * hnn);
    float hNew = (1.0f - z) * n + z * h[i];
    h[i] = hNew;
    h16[i] = (_Float16)hNew;
    storage[i] = hNew;
}

// ---------------- pred = h@W_fc + b_fc, log_softmax, loss accumulate ----------
// One block; warp lane == vocab index (wave32 == VOC_). Single writer on loss.

__global__ __launch_bounds__(1024) void k_pred_loss(
    const float* __restrict__ h, const float* __restrict__ Wfc,
    const float* __restrict__ bfc, const int* __restrict__ tgt,
    float* __restrict__ preds, float* __restrict__ lossPtr, int t)
{
    __shared__ float lpart[B_];
    const int lane = threadIdx.x & 31;
    const int w = threadIdx.x >> 5;                // 0..31
    for (int b = w; b < B_; b += 32) {
        float acc = bfc[lane];
        const float* hb = h + (size_t)b * HID_;
        for (int k = 0; k < HID_; ++k)
            acc += hb[k] * Wfc[k * VOC_ + lane];
        preds[b * VOC_ + lane] = acc;
        float m = acc;
        for (int off = 16; off; off >>= 1) m = fmaxf(m, __shfl_xor(m, off, 32));
        float e = __expf(acc - m);
        float s = e;
        for (int off = 16; off; off >>= 1) s += __shfl_xor(s, off, 32);
        float logp = acc - m - __logf(s);
        if (lane == tgt[b * T_ + t]) lpart[b] = logp;
    }
    __syncthreads();
    if (threadIdx.x == 0) {
        float sum = 0.0f;
        for (int b = 0; b < B_; ++b) sum += lpart[b];
        *lossPtr += -sum * (1.0f / (float)B_);
    }
}

// ---------------- host side ---------------------------------------------------

extern "C" void kernel_launch(void* const* d_in, const int* in_sizes, int n_in,
                              void* d_out, int out_size, void* d_ws, size_t ws_size,
                              hipStream_t stream) {
    const float* Ef   = (const float*)d_in[0];   // (64,256,512)
    const int*   tgt  = (const int*)  d_in[1];   // (64,100)
    const float* emb  = (const float*)d_in[2];   // (32,256)
    const float* We   = (const float*)d_in[3];   // (512,512)
    const float* be   = (const float*)d_in[4];
    const float* Wh   = (const float*)d_in[5];   // (512,512)
    const float* bh   = (const float*)d_in[6];
    const float* vatt = (const float*)d_in[7];
    const float* Wih  = (const float*)d_in[8];   // (1536,768)
    const float* Whh  = (const float*)d_in[9];   // (1536,512)
    const float* bih  = (const float*)d_in[10];
    const float* bhh  = (const float*)d_in[11];
    const float* Wfc  = (const float*)d_in[12];  // (512,32)
    const float* bfc  = (const float*)d_in[13];

    float* out = (float*)d_out;
    const size_t PRED_N    = (size_t)T_ * B_ * VOC_;   // 204800
    const size_t LOSS_IDX  = PRED_N;
    const size_t STORE_OFF = PRED_N + 1;

    // workspace carve-up (256B aligned)
    char* wsb = (char*)d_ws;
    size_t off = 0;
    auto carve = [&](size_t bytes) -> void* {
        void* p = wsb + off;
        off = (off + bytes + 255) & ~(size_t)255;
        return p;
    };
    const size_t EP = (size_t)B_ * P_;                     // 16384 rows
    float*    encp  = (float*)   carve(EP * ATT_ * 4);     // 33.5 MB
    _Float16* Ef16  = (_Float16*)carve(EP * ENC_ * 2);     // 16.8 MB
    _Float16* WeT   = (_Float16*)carve((size_t)ATT_ * ENC_ * 2);
    _Float16* WhT   = (_Float16*)carve((size_t)ATT_ * HID_ * 2);
    _Float16* Wih16 = (_Float16*)carve((size_t)3 * HID_ * (ENC_ + EMB_) * 2);
    _Float16* Whh16 = (_Float16*)carve((size_t)3 * HID_ * HID_ * 2);
    _Float16* emb16 = (_Float16*)carve((size_t)VOC_ * EMB_ * 2);
    float*    g     = (float*)   carve((size_t)B_ * ATT_ * 4);
    float*    sc    = (float*)   carve((size_t)B_ * P_ * 4);
    _Float16* x16   = (_Float16*)carve((size_t)B_ * (ENC_ + EMB_) * 2);
    float*    h     = (float*)   carve((size_t)B_ * HID_ * 4);
    _Float16* h16   = (_Float16*)carve((size_t)B_ * HID_ * 2);
    float*    gi    = (float*)   carve((size_t)B_ * 3 * HID_ * 4);
    float*    gh    = (float*)   carve((size_t)B_ * 3 * HID_ * 4);
    (void)ws_size; (void)in_sizes; (void)n_in; (void)out_size;

    // ---- one-time setup (per call, deterministic) ----
    int nEf = (int)(EP * ENC_);
    k_convert_f16  <<<(nEf + 255) / 256, 256, 0, stream>>>(Ef, Ef16, nEf);
    k_transpose_f16<<<(ENC_ * ATT_ + 255) / 256, 256, 0, stream>>>(We, WeT, ENC_, ATT_);
    k_transpose_f16<<<(HID_ * ATT_ + 255) / 256, 256, 0, stream>>>(Wh, WhT, HID_, ATT_);
    k_convert_f16  <<<(3 * HID_ * (ENC_ + EMB_) + 255) / 256, 256, 0, stream>>>(Wih, Wih16, 3 * HID_ * (ENC_ + EMB_));
    k_convert_f16  <<<(3 * HID_ * HID_ + 255) / 256, 256, 0, stream>>>(Whh, Whh16, 3 * HID_ * HID_);
    k_convert_f16  <<<(VOC_ * EMB_ + 255) / 256, 256, 0, stream>>>(emb, emb16, VOC_ * EMB_);
    k_init<<<(B_ * HID_ + 255) / 256, 256, 0, stream>>>(h, h16, out + LOSS_IDX);

    // enc_proj = Ef @ We + be : 16384x512x512 via WMMA (32768 tiles / 8 waves)
    k_wmma_gemm<<<4096, 256, 0, stream>>>(Ef16, WeT, be, encp, (int)EP, ATT_, ENC_);

    // ---- 100 sequential decode steps ----
    for (int t = 0; t < T_; ++t) {
        // g = h @ Wh + bh : 64x512x512 (128 tiles)
        k_wmma_gemm<<<16, 256, 0, stream>>>(h16, WhT, bh, g, B_, ATT_, HID_);
        k_att_scores<<<B_, 256, 0, stream>>>(encp, g, vatt, sc);
        k_softmax_ctx<<<B_, 256, 0, stream>>>(sc, Ef, emb16, tgt, x16, t);
        // gi = x @ W_ih^T + b_ih : 64x1536x768 (384 tiles)
        k_wmma_gemm<<<48, 256, 0, stream>>>(x16, Wih16, bih, gi, B_, 3 * HID_, ENC_ + EMB_);
        // gh = h @ W_hh^T + b_hh : 64x1536x512 (384 tiles)  [old h16]
        k_wmma_gemm<<<48, 256, 0, stream>>>(h16, Whh16, bhh, gh, B_, 3 * HID_, HID_);
        k_gru<<<(B_ * HID_) / 256, 256, 0, stream>>>(gi, gh, h, h16,
                                                     out + STORE_OFF + (size_t)t * B_ * HID_);
        k_pred_loss<<<1, 1024, 0, stream>>>(h, Wfc, bfc, tgt,
                                            out + (size_t)t * B_ * VOC_,
                                            out + LOSS_IDX, t);
    }
}
